// EpisodicMemory_10058813407234
// MI455X (gfx1250) — compile-verified, matrix operand order checked by reading, not run on hardware
//
#include <hip/hip_runtime.h>
#include <hip/hip_bf16.h>
#include <math.h>

#define Hdim 1024
#define Sdim 128
#define Bdim 128
#define KCHUNK 128

typedef __attribute__((ext_vector_type(16))) __bf16        v16bf;
typedef __attribute__((ext_vector_type(8)))  float         v8f;
typedef __attribute__((ext_vector_type(4)))  unsigned int  u32x4;
typedef __attribute__((ext_vector_type(8)))  int           i32x8;
typedef __attribute__((ext_vector_type(4)))  int           i32x4;

// ---------------------------------------------------------------------------
// CDNA5 bf16 WMMA: D(16x16,f32) = A(16x32,bf16) x B(32x16,bf16) + C
// A lane layout (wave32): lane = 16*half + m ; element e holds K = (e&7)+8*half+16*(e>>3)
// i.e. per lane: floats [base+0..7] and [base+16..23] with base = ktile + 8*half.
// ---------------------------------------------------------------------------
__device__ __forceinline__ v8f wmma_bf16(v16bf a, v16bf b, v8f c) {
    return __builtin_amdgcn_wmma_f32_16x16x32_bf16(
        /*neg_a=*/false, a, /*neg_b=*/false, b,
        /*c_mod=*/(short)0, c, /*reuse_a=*/false, /*reuse_b=*/false);
}

// Load 16 f32 (two contiguous 8-float runs: [0..7], [16..23]) -> bf16 fragment.
// Works for global or LDS pointers (AS inferred after inlining).
__device__ __forceinline__ v16bf frag16_g(const float* base) {
    const float4* p = (const float4*)base;
    float4 c0 = p[0], c1 = p[1], c2 = p[4], c3 = p[5];
    v16bf r;
    r[0]=(__bf16)c0.x;  r[1]=(__bf16)c0.y;  r[2]=(__bf16)c0.z;  r[3]=(__bf16)c0.w;
    r[4]=(__bf16)c1.x;  r[5]=(__bf16)c1.y;  r[6]=(__bf16)c1.z;  r[7]=(__bf16)c1.w;
    r[8]=(__bf16)c2.x;  r[9]=(__bf16)c2.y;  r[10]=(__bf16)c2.z; r[11]=(__bf16)c2.w;
    r[12]=(__bf16)c3.x; r[13]=(__bf16)c3.y; r[14]=(__bf16)c3.z; r[15]=(__bf16)c3.w;
    return r;
}

// ---------------------------------------------------------------------------
// Tensor Data Mover: DMA a 2D f32 tile (tile_rows x tile_k, row stride
// row_stride_elems) from global into LDS, row-major compact. D# per ISA
// cdna5 8.3/8.4: group0 {count=1 | lds_addr | global_addr | type=2},
// group1 {data_size=4B, tensor dims == tile dims (no OOB), tile dims, stride}.
// Tracked with TENSORcnt. 6-arg builtin (clang-23 / therock toolchain).
// ---------------------------------------------------------------------------
__device__ __forceinline__ void tdm_load_2d(const float* gsrc, float* lds_dst,
                                            unsigned tile_k, unsigned tile_rows,
                                            unsigned row_stride_elems)
{
    unsigned long long ga = (unsigned long long)(uintptr_t)gsrc;
    unsigned lo = (unsigned)(uintptr_t)lds_dst;          // LDS byte offset (addr[31:0])
    u32x4 g0 = { 1u,                                     // count=1, user descriptor
                 lo,                                     // lds_addr [63:32]
                 (unsigned)ga,                           // global_addr [95:64]
                 (unsigned)((ga >> 32) & 0x01FFFFFFu) | 0x80000000u }; // ga hi + type=2
    i32x8 g1 = { (int)(2u << 16),                        // data_size = 4 bytes
                 (int)(tile_k << 16),                    // tensor_dim0[15:0] @ [63:48]
                 (int)((tile_k >> 16) | (tile_rows << 16)),   // td0 hi | td1 lo
                 (int)((tile_rows >> 16) | (tile_k << 16)),   // td1 hi | tile_dim0
                 (int)(tile_rows & 0xFFFFu),             // tile_dim1 (tile_dim2 = 0)
                 (int)row_stride_elems,                  // tensor_dim0_stride[31:0]
                 0, 0 };
    i32x4 gz4 = { 0, 0, 0, 0 };                          // groups 2/3 unused (2D)
    i32x8 gz8 = { 0, 0, 0, 0, 0, 0, 0, 0 };
    __builtin_amdgcn_tensor_load_to_lds(g0, g1, gz4, gz4, gz8, 0);
}

// Build A-fragment of z = [f*q | f*m | |f-q| | |f-m|] on the fly.
__device__ __forceinline__ v16bf make_z_frag(const float* __restrict__ fbase,
                                             const float* oth, int j, int seg) {
    float4 f0 = *(const float4*)(fbase + j);
    float4 f1 = *(const float4*)(fbase + j + 4);
    float4 f2 = *(const float4*)(fbase + j + 16);
    float4 f3 = *(const float4*)(fbase + j + 20);
    float4 o0 = *(const float4*)(oth + j);
    float4 o1 = *(const float4*)(oth + j + 4);
    float4 o2 = *(const float4*)(oth + j + 16);
    float4 o3 = *(const float4*)(oth + j + 20);
    float fv[16] = {f0.x,f0.y,f0.z,f0.w, f1.x,f1.y,f1.z,f1.w,
                    f2.x,f2.y,f2.z,f2.w, f3.x,f3.y,f3.z,f3.w};
    float ov[16] = {o0.x,o0.y,o0.z,o0.w, o1.x,o1.y,o1.z,o1.w,
                    o2.x,o2.y,o2.z,o2.w, o3.x,o3.y,o3.z,o3.w};
    bool mul = (seg < 2);
    v16bf r;
#pragma unroll
    for (int e = 0; e < 16; ++e) {
        float z = mul ? (fv[e] * ov[e]) : fabsf(fv[e] - ov[e]);
        r[e] = (__bf16)z;
    }
    return r;
}

// ---------------------------------------------------------------------------
// Kernel 1: attention-gate MLP. G_logit[b,s] = Wz2 . tanh(z @ Wz1^T + bz1) + bz2
// WG = 16 rows (one b, 16 s). Outer loop: 8 n-blocks of 128 columns; Wz1 tile
// (128 rows x KCHUNK) TDM-staged into LDS and shared by all 8 waves. tanh+dot
// reduction fused via LDS float atomics.
// ---------------------------------------------------------------------------
__global__ __launch_bounds__(256) void gate_kernel(
    const float* __restrict__ facts, const float* __restrict__ q,
    const float* __restrict__ pm,    const float* __restrict__ Wz1,
    const float* __restrict__ bz1,   const float* __restrict__ Wz2,
    const float* __restrict__ bz2,   float* __restrict__ Glogit)
{
    __shared__ float qs[Hdim];
    __shared__ float ms[Hdim];
    __shared__ float wstage[128 * KCHUNK];     // 64 KB staged Wz1 tile
    __shared__ float rowsum[16];

    int b  = blockIdx.x >> 3;
    int s0 = (blockIdx.x & 7) * 16;
    int tid = threadIdx.x;

    for (int i = tid; i < Hdim; i += 256) {
        qs[i] = q [(size_t)b * Hdim + i];
        ms[i] = pm[(size_t)b * Hdim + i];
    }
    if (tid < 16) rowsum[tid] = 0.f;
    __syncthreads();

    int wave = tid >> 5, lane = tid & 31, half = lane >> 4, l16 = lane & 15;
    const float* fbase = facts + ((size_t)(b * Sdim + s0 + l16)) * Hdim;

    float psum[8];
#pragma unroll
    for (int r = 0; r < 8; ++r) psum[r] = 0.f;

    for (int nb = 0; nb < 8; ++nb) {                     // 128-column n-block
        int n = nb * 128 + wave * 16 + l16;
        v8f acc;
        float bias = bz1[n];
#pragma unroll
        for (int r = 0; r < 8; ++r) acc[r] = bias;

        for (int kc = 0; kc < 4 * Hdim; kc += KCHUNK) {
            if (wave == 0) {                             // one TDM per chunk
                tdm_load_2d(Wz1 + (size_t)(nb * 128) * (4 * Hdim) + kc,
                            wstage, KCHUNK, 128, 4 * Hdim);
                __builtin_amdgcn_s_wait_tensorcnt(0);
            }
            __syncthreads();
#pragma unroll
            for (int kk = 0; kk < KCHUNK; kk += 32) {
                int kb  = kc + kk;
                int seg = kb >> 10;
                int j   = (kb & (Hdim - 1)) + 8 * half;
                const float* oth = (seg & 1) ? ms : qs;  // seg 0,2 -> q ; 1,3 -> m
                v16bf a  = make_z_frag(fbase, oth, j, seg);
                v16bf bf = frag16_g(&wstage[(wave * 16 + l16) * KCHUNK + kk + 8 * half]);
                acc = wmma_bf16(a, bf, acc);
            }
            __syncthreads();                             // before TDM overwrites
        }
        float w2 = Wz2[n];
#pragma unroll
        for (int r = 0; r < 8; ++r) psum[r] += tanhf(acc[r]) * w2;
    }
#pragma unroll
    for (int r = 0; r < 8; ++r) atomicAdd(&rowsum[r + 8 * half], psum[r]); // ds_add_f32
    __syncthreads();
    if (tid < 16) Glogit[(size_t)b * Sdim + s0 + tid] = rowsum[tid] + bz2[0];
}

// ---------------------------------------------------------------------------
// Kernel 2: softmax over S per batch (in place).
// ---------------------------------------------------------------------------
__global__ __launch_bounds__(128) void softmax_kernel(float* __restrict__ Gio)
{
    __shared__ float red[128];
    int b = blockIdx.x, t = threadIdx.x;
    float x = Gio[(size_t)b * Sdim + t];
    red[t] = x;
    __syncthreads();
    for (int off = 64; off > 0; off >>= 1) {
        if (t < off) red[t] = fmaxf(red[t], red[t + off]);
        __syncthreads();
    }
    float mx = red[0];
    __syncthreads();
    float e = __expf(x - mx);
    red[t] = e;
    __syncthreads();
    for (int off = 64; off > 0; off >>= 1) {
        if (t < off) red[t] += red[t + off];
        __syncthreads();
    }
    Gio[(size_t)b * Sdim + t] = e / red[0];
}

__global__ void zero_kernel(float* __restrict__ p, int n)
{
    int i = blockIdx.x * blockDim.x + threadIdx.x;
    if (i < n) p[i] = 0.f;
}

// ---------------------------------------------------------------------------
// Kernel 3: one GRU-style scan step (launched 128x; host loop = global sync).
// WG owns 16 output columns; waves own the 8 m-tiles. Weight rows for all four
// matrices are TDM-staged per 128-wide K-chunk and shared by all waves.
// ---------------------------------------------------------------------------
__global__ __launch_bounds__(256) void step_kernel(
    const float* __restrict__ facts, const float* __restrict__ G,
    const float* __restrict__ Cin,   float* __restrict__ Cout,
    const float* __restrict__ Wr,    const float* __restrict__ br,
    const float* __restrict__ Ur,    const float* __restrict__ bur,
    const float* __restrict__ W,     const float* __restrict__ bw,
    const float* __restrict__ U,     const float* __restrict__ bu, int s)
{
    __shared__ float wS[4][16 * KCHUNK];                 // 32 KB: Wr/Ur/W/U tiles

    int tid = threadIdx.x, wave = tid >> 5, lane = tid & 31;
    int half = lane >> 4, l16 = lane & 15;
    int n0 = blockIdx.x * 16;                            // 64 WGs cover N=1024
    int n  = n0 + l16;
    int arow = wave * 16 + l16;                          // batch row fed as A

    const float* fA = facts + ((size_t)arow * Sdim + s) * Hdim;
    const float* cA = Cin + (size_t)arow * Hdim;

    v8f aR = {}, aUr = {}, aW = {}, aU = {};
    for (int kc = 0; kc < Hdim; kc += KCHUNK) {
        if (wave == 0) {
            tdm_load_2d(Wr + (size_t)n0 * Hdim + kc, wS[0], KCHUNK, 16, Hdim);
            tdm_load_2d(Ur + (size_t)n0 * Hdim + kc, wS[1], KCHUNK, 16, Hdim);
            tdm_load_2d(W  + (size_t)n0 * Hdim + kc, wS[2], KCHUNK, 16, Hdim);
            tdm_load_2d(U  + (size_t)n0 * Hdim + kc, wS[3], KCHUNK, 16, Hdim);
            __builtin_amdgcn_s_wait_tensorcnt(0);
        }
        __syncthreads();
        if (kc + KCHUNK < Hdim) {                        // global_prefetch_b8 (A stream)
            __builtin_prefetch(fA + kc + KCHUNK, 0, 1);
            __builtin_prefetch(cA + kc + KCHUNK, 0, 1);
        }
#pragma unroll
        for (int kk = 0; kk < KCHUNK; kk += 32) {
            int o  = kc + kk + 8 * half;
            int lo = l16 * KCHUNK + kk + 8 * half;
            v16bf af = frag16_g(fA + o);
            v16bf ac = frag16_g(cA + o);
            aR  = wmma_bf16(af, frag16_g(&wS[0][lo]), aR);
            aUr = wmma_bf16(ac, frag16_g(&wS[1][lo]), aUr);
            aW  = wmma_bf16(af, frag16_g(&wS[2][lo]), aW);
            aU  = wmma_bf16(ac, frag16_g(&wS[3][lo]), aU);
        }
        __syncthreads();
    }

    float brn = br[n] + bur[n];
    float bwn = bw[n], bun = bu[n];
#pragma unroll
    for (int r = 0; r < 8; ++r) {
        int m  = wave * 16 + r + 8 * half;               // batch index of D element
        float g  = G[(size_t)m * Sdim + s];
        float c  = Cin[(size_t)m * Hdim + n];
        float rg = 1.f / (1.f + __expf(-(aR[r] + aUr[r] + brn)));
        float ht = tanhf(aW[r] + bwn + rg * (aU[r] + bun));
        Cout[(size_t)m * Hdim + n] = g * ht + (1.f - g) * c;
    }
}

// ---------------------------------------------------------------------------
// Kernel 4: next_mem = relu([prevM | C | q] @ Wm^T + bm). Same tiling as step.
// ---------------------------------------------------------------------------
__global__ __launch_bounds__(256) void final_kernel(
    const float* __restrict__ pm,  const float* __restrict__ C,
    const float* __restrict__ qst, const float* __restrict__ Wm,
    const float* __restrict__ bm,  float* __restrict__ out)
{
    __shared__ float wS[16 * KCHUNK];                    // 8 KB staged Wm tile

    int tid = threadIdx.x, wave = tid >> 5, lane = tid & 31;
    int half = lane >> 4, l16 = lane & 15;
    int n0 = blockIdx.x * 16;
    int n  = n0 + l16;
    int brow = wave * 16 + l16;

    const float* a0 = pm  + (size_t)brow * Hdim;
    const float* a1 = C   + (size_t)brow * Hdim;
    const float* a2 = qst + (size_t)brow * Hdim;

    v8f acc = {};
    for (int kc = 0; kc < 3 * Hdim; kc += KCHUNK) {
        if (wave == 0) {
            tdm_load_2d(Wm + (size_t)n0 * (3 * Hdim) + kc, wS, KCHUNK, 16, 3 * Hdim);
            __builtin_amdgcn_s_wait_tensorcnt(0);
        }
        __syncthreads();
        int seg = kc >> 10;
        const float* ar = (seg == 0) ? a0 : (seg == 1) ? a1 : a2;
#pragma unroll
        for (int kk = 0; kk < KCHUNK; kk += 32) {
            v16bf a  = frag16_g(ar + ((kc + kk) & (Hdim - 1)) + 8 * half);
            v16bf bf = frag16_g(&wS[l16 * KCHUNK + kk + 8 * half]);
            acc = wmma_bf16(a, bf, acc);
        }
        __syncthreads();
    }
    float bias = bm[n];
#pragma unroll
    for (int r = 0; r < 8; ++r) {
        int m = wave * 16 + r + 8 * half;
        float v = acc[r] + bias;
        out[(size_t)m * Hdim + n] = v > 0.f ? v : 0.f;
    }
}

// ---------------------------------------------------------------------------
extern "C" void kernel_launch(void* const* d_in, const int* in_sizes, int n_in,
                              void* d_out, int out_size, void* d_ws, size_t ws_size,
                              hipStream_t stream)
{
    (void)in_sizes; (void)n_in; (void)out_size; (void)ws_size;
    const float* facts = (const float*)d_in[0];
    const float* ques  = (const float*)d_in[1];
    const float* prevM = (const float*)d_in[2];
    const float* Wr  = (const float*)d_in[3];
    const float* br  = (const float*)d_in[4];
    const float* Ur  = (const float*)d_in[5];
    const float* bur = (const float*)d_in[6];
    const float* W   = (const float*)d_in[7];
    const float* bw  = (const float*)d_in[8];
    const float* U   = (const float*)d_in[9];
    const float* bu  = (const float*)d_in[10];
    const float* Wz1 = (const float*)d_in[11];
    const float* bz1 = (const float*)d_in[12];
    const float* Wz2 = (const float*)d_in[13];
    const float* bz2 = (const float*)d_in[14];
    const float* Wm  = (const float*)d_in[15];
    const float* bm  = (const float*)d_in[16];

    float* ws = (float*)d_ws;
    float* G  = ws;                         // B*S = 16384 f32
    float* Ca = ws + 16384;                 // B*H = 131072 f32
    float* Cb = Ca + (Bdim * Hdim);         // B*H = 131072 f32

    gate_kernel<<<1024, 256, 0, stream>>>(facts, ques, prevM, Wz1, bz1, Wz2, bz2, G);
    softmax_kernel<<<Bdim, Sdim, 0, stream>>>(G);
    zero_kernel<<<(Bdim * Hdim + 255) / 256, 256, 0, stream>>>(Ca, Bdim * Hdim);

    float* cin = Ca, *cout = Cb;
    for (int s = 0; s < Sdim; ++s) {
        step_kernel<<<64, 256, 0, stream>>>(facts, G, cin, cout,
                                            Wr, br, Ur, bur, W, bw, U, bu, s);
        float* tmp = cin; cin = cout; cout = tmp;
    }

    final_kernel<<<64, 256, 0, stream>>>(prevM, cin, ques, Wm, bm, (float*)d_out);
}
